// BiasMultiheadAttention_773094113359
// MI455X (gfx1250) — matrix operand order repaired and checked by reading, not compiled, not run on hardware
//
#include <hip/hip_runtime.h>
#include <hip/hip_bf16.h>

// ---------------------------------------------------------------------------
// BiasMultiheadAttention for MI455X (gfx1250): bf16 WMMA + async-to-LDS pipeline
//   B=4, S=2048, D=512, H=8, Dh=64
// ---------------------------------------------------------------------------

typedef __bf16  v16bf __attribute__((ext_vector_type(16)));
typedef float   v8f   __attribute__((ext_vector_type(8)));
typedef int     v4i_a __attribute__((ext_vector_type(4)));

#define NB 4
#define NS 2048
#define ND 512
#define NH 8
#define NDH 64
#define N3D 1536
#define NTOK (NB * NS)          // 8192
constexpr float kScale = 0.125f; // 1/sqrt(64)

// --------------------------- async global -> LDS ---------------------------
#if defined(__gfx1250__) && __has_builtin(__builtin_amdgcn_global_load_async_to_lds_b128)
#define HAVE_ASYNC 1
#else
#define HAVE_ASYNC 0
#endif

// Copy 16 bytes global -> LDS (async on gfx1250 when available).
static __device__ __forceinline__ void cp16(__bf16* lds, const __bf16* g) {
#if HAVE_ASYNC
    __builtin_amdgcn_global_load_async_to_lds_b128(
        (v4i_a*)g,
        (__attribute__((address_space(3))) v4i_a*)lds, 0, 0);
#else
    *(uint4*)lds = *(const uint4*)g;
#endif
}

static __device__ __forceinline__ void async_wait() {
#if HAVE_ASYNC
#if __has_builtin(__builtin_amdgcn_s_wait_asynccnt)
    __builtin_amdgcn_s_wait_asynccnt(0);
#else
    asm volatile("s_wait_asynccnt 0" ::: "memory");
#endif
#endif
}

// Build a 32-byte WMMA fragment from two contiguous 16B LDS chunks.
static __device__ __forceinline__ v16bf load_frag(const __bf16* p0, const __bf16* p1) {
    union { v16bf v; uint4 q[2]; } u;
    u.q[0] = *(const uint4*)p0;
    u.q[1] = *(const uint4*)p1;
    return u.v;
}

// ---------------------------------------------------------------------------
// 1) fp32 -> bf16 conversion
// ---------------------------------------------------------------------------
__global__ void cvt_bf16_kernel(const float* __restrict__ s, __bf16* __restrict__ d, int n) {
    int i = blockIdx.x * blockDim.x + threadIdx.x;
    int stride = gridDim.x * blockDim.x;
    for (; i < n; i += stride) d[i] = (__bf16)s[i];
}

// ---------------------------------------------------------------------------
// 2) Fused QKV GEMM:  qkv[t, o] = sum_d x[t,d] * w_in[o,d] + b_in[o]
//    Block tile 128(M) x 64(N), K-step 32, double-buffered async LDS staging.
//    Scatter result into Q/K/V [B,H,S,Dh] (bf16).
// ---------------------------------------------------------------------------
__global__ __launch_bounds__(256) void qkv_gemm_kernel(
        const __bf16* __restrict__ X, const __bf16* __restrict__ W,
        const float* __restrict__ bias,
        __bf16* __restrict__ Qo, __bf16* __restrict__ Ko, __bf16* __restrict__ Vo) {
    __shared__ __bf16 As[2][128][40];  // [m][k], pad 8
    __shared__ __bf16 Bt[2][64][40];   // [n][k], pad 8 (N-major so frags are contiguous)

    const int bm = blockIdx.x * 128, bn = blockIdx.y * 64;
    const int tid = threadIdx.x;
    const int w = tid >> 5, lane = tid & 31;
    const int g = lane >> 4, cn = lane & 15;
    const int wr = w >> 1, wc = w & 1;   // wave covers rows [wr*32,+32), cols [wc*32,+32)

    // staging coordinates: 16B chunks
    const int sr = tid >> 2, ss = (tid & 3) * 8;

    v8f acc[2][2] = {};

    // prologue: stage tile 0
    cp16(&As[0][sr][ss],      &X[(size_t)(bm + sr) * ND + ss]);
    cp16(&As[0][sr + 64][ss], &X[(size_t)(bm + sr + 64) * ND + ss]);
    cp16(&Bt[0][sr][ss],      &W[(size_t)(bn + sr) * ND + ss]);
    async_wait();
    __syncthreads();

    for (int k0 = 0; k0 < ND; k0 += 32) {
        const int buf = (k0 >> 5) & 1;
        if (k0 + 32 < ND) {
            const int nk = k0 + 32, nb = buf ^ 1;
            cp16(&As[nb][sr][ss],      &X[(size_t)(bm + sr) * ND + nk + ss]);
            cp16(&As[nb][sr + 64][ss], &X[(size_t)(bm + sr + 64) * ND + nk + ss]);
            cp16(&Bt[nb][sr][ss],      &W[(size_t)(bn + sr) * ND + nk + ss]);
        }

        v16bf a[2], b[2];
        for (int i = 0; i < 2; i++) {
            int row = wr * 32 + i * 16 + cn;
            a[i] = load_frag(&As[buf][row][8 * g], &As[buf][row][16 + 8 * g]);
        }
        for (int j = 0; j < 2; j++) {
            int col = wc * 32 + j * 16 + cn;
            b[j] = load_frag(&Bt[buf][col][8 * g], &Bt[buf][col][16 + 8 * g]);
        }
        for (int i = 0; i < 2; i++)
            for (int j = 0; j < 2; j++)
                acc[i][j] = __builtin_amdgcn_wmma_f32_16x16x32_bf16(
                    false, a[i], false, b[j], (short)0, acc[i][j], false, false);

        async_wait();
        __syncthreads();
    }

    // epilogue: + b_in, scatter into Q/K/V [B,H,S,Dh]
    for (int i = 0; i < 2; i++)
        for (int j = 0; j < 2; j++) {
            int gcol_base = bn + wc * 32 + j * 16;    // multiple of 16
            int which = gcol_base >> 9;               // 0=Q 1=K 2=V (uniform per tile)
            int o = gcol_base & 511;
            int h = o >> 6;
            int dh0 = o & 63;                          // multiple of 16
            __bf16* dst = (which == 0) ? Qo : ((which == 1) ? Ko : Vo);
            float bb = bias[gcol_base + cn];
            for (int r = 0; r < 8; r++) {
                int grow = bm + wr * 32 + i * 16 + r + 8 * g;
                int b_ = grow >> 11, s_ = grow & 2047;
                dst[(size_t)(((b_ * NH + h) * NS) + s_) * NDH + dh0 + cn] =
                    (__bf16)(acc[i][j][r] + bb);
            }
        }
}

// ---------------------------------------------------------------------------
// 3) Flash attention per (b, h, 64-row q tile), streaming 64-wide K/V chunks
//    with double-buffered async staging. Scores f32 in LDS, online softmax,
//    P bf16, P@V via WMMA into persistent per-wave f32 accumulators.
// ---------------------------------------------------------------------------
__global__ __launch_bounds__(256) void attn_kernel(
        const __bf16* __restrict__ Q, const __bf16* __restrict__ K,
        const __bf16* __restrict__ V, const float* __restrict__ bias,
        __bf16* __restrict__ O) {
    const int qt = blockIdx.x;        // q-tile index 0..31
    const int h  = blockIdx.y;
    const int b  = blockIdx.z;

    __shared__ __bf16 Qs[64][72];
    __shared__ __bf16 Ks[2][64][72];  // K chunk row-major == Bt layout for Q@K^T
    __shared__ __bf16 Vr[2][64][72];  // V chunk row-major (async staged)
    __shared__ __bf16 Vt[64][72];     // V transposed: Vt[dh][sk] == Bt layout for P@V
    __shared__ __bf16 Ps[64][72];     // probabilities (bf16)
    __shared__ float  Sc[64][68];     // raw scores (f32)
    __shared__ float  mrow[64], lrow[64], arow[64], red[64][4];

    const int tid = threadIdx.x;
    const int w = tid >> 5, lane = tid & 31;
    const int g = lane >> 4, cn = lane & 15;
    const int mi = w >> 1, nc = w & 1;  // wave: m-tile mi, n-tiles {2nc, 2nc+1}

    const __bf16* Qbase = Q + (size_t)((b * NH + h) * NS + qt * 64) * NDH;
    const __bf16* Kbase = K + (size_t)((b * NH + h) * NS) * NDH;
    const __bf16* Vbase = V + (size_t)((b * NH + h) * NS) * NDH;
    const float*  brow  = bias + (size_t)(h * NS + qt * 64) * NS;

    // staging coordinates: two 16B chunks per thread per 64x64 tile
    const int sr = tid >> 3, ss = (tid & 7) * 8;

    // prologue: stage Q tile + chunk 0 of K/V
    cp16(&Qs[sr][ss],      &Qbase[(size_t)sr * NDH + ss]);
    cp16(&Qs[sr + 32][ss], &Qbase[(size_t)(sr + 32) * NDH + ss]);
    cp16(&Ks[0][sr][ss],      &Kbase[(size_t)sr * NDH + ss]);
    cp16(&Ks[0][sr + 32][ss], &Kbase[(size_t)(sr + 32) * NDH + ss]);
    cp16(&Vr[0][sr][ss],      &Vbase[(size_t)sr * NDH + ss]);
    cp16(&Vr[0][sr + 32][ss], &Vbase[(size_t)(sr + 32) * NDH + ss]);
    if (tid < 64) { mrow[tid] = -3.0e38f; lrow[tid] = 0.f; }
    async_wait();
    __syncthreads();

    // preload Q A-fragments (constant over chunks): 2 K-steps of 32
    v16bf qa[2];
    {
        int row = mi * 16 + cn;
        for (int ks = 0; ks < 2; ks++)
            qa[ks] = load_frag(&Qs[row][ks * 32 + 8 * g], &Qs[row][ks * 32 + 16 + 8 * g]);
    }

    v8f oacc[2] = {};

    for (int kc = 0; kc < NS; kc += 64) {
        const int buf = (kc >> 6) & 1;

        // issue async staging of the next chunk into the other buffer
        if (kc + 64 < NS) {
            const int nb = buf ^ 1, nk = kc + 64;
            cp16(&Ks[nb][sr][ss],      &Kbase[(size_t)(nk + sr) * NDH + ss]);
            cp16(&Ks[nb][sr + 32][ss], &Kbase[(size_t)(nk + sr + 32) * NDH + ss]);
            cp16(&Vr[nb][sr][ss],      &Vbase[(size_t)(nk + sr) * NDH + ss]);
            cp16(&Vr[nb][sr + 32][ss], &Vbase[(size_t)(nk + sr + 32) * NDH + ss]);
        }

        // transpose V chunk: Vr[buf][sk][dh] -> Vt[dh][sk]
        for (int half = 0; half < 2; half++) {
            int r = sr + half * 32;
            union { uint4 q; __bf16 e[8]; } u;
            u.q = *(const uint4*)&Vr[buf][r][ss];
            for (int e = 0; e < 8; e++) Vt[ss + e][r] = u.e[e];
        }

        // S = Q @ K^T : wave computes tiles (mi, 2nc) and (mi, 2nc+1)
        v8f sacc[2] = {};
        for (int ks = 0; ks < 2; ks++)
            for (int j = 0; j < 2; j++) {
                int col = (nc * 2 + j) * 16 + cn;
                v16bf bk = load_frag(&Ks[buf][col][ks * 32 + 8 * g],
                                     &Ks[buf][col][ks * 32 + 16 + 8 * g]);
                sacc[j] = __builtin_amdgcn_wmma_f32_16x16x32_bf16(
                    false, qa[ks], false, bk, (short)0, sacc[j], false, false);
            }
        for (int j = 0; j < 2; j++) {
            int col = (nc * 2 + j) * 16 + cn;
            for (int r = 0; r < 8; r++)
                Sc[mi * 16 + r + 8 * g][col] = sacc[j][r];
        }
        __syncthreads();   // Sc + Vt ready

        // online softmax: 4 threads per row, 16 cols each
        {
            int row = tid >> 2, quad = tid & 3;
            const float* bptr = brow + (size_t)row * NS + kc + quad * 16;
            if (kc + 64 < NS) __builtin_prefetch(bptr + 64, 0, 0);
            float sv[16];
            float vmax = -3.0e38f;
            for (int c = 0; c < 16; c++) {
                float s = Sc[row][quad * 16 + c] * kScale + bptr[c];
                sv[c] = s;
                vmax = fmaxf(vmax, s);
            }
            red[row][quad] = vmax;
            __syncthreads();
            if (quad == 0) {
                float cm = fmaxf(fmaxf(red[row][0], red[row][1]),
                                 fmaxf(red[row][2], red[row][3]));
                float mn = fmaxf(mrow[row], cm);
                arow[row] = __expf(mrow[row] - mn);
                mrow[row] = mn;
            }
            __syncthreads();
            float mn = mrow[row];
            float psum = 0.f;
            for (int c = 0; c < 16; c++) {
                float p = __expf(sv[c] - mn);
                psum += p;
                Ps[row][quad * 16 + c] = (__bf16)p;
            }
            red[row][quad] = psum;
            __syncthreads();
            if (quad == 0)
                lrow[row] = lrow[row] * arow[row] +
                            (red[row][0] + red[row][1] + red[row][2] + red[row][3]);
        }
        __syncthreads();   // Ps ready, arow ready for all waves

        // rescale running O by alpha, then O += P @ V
        for (int j = 0; j < 2; j++)
            for (int r = 0; r < 8; r++)
                oacc[j][r] *= arow[mi * 16 + r + 8 * g];

        for (int ks = 0; ks < 2; ks++) {
            int prow = mi * 16 + cn;
            v16bf pa = load_frag(&Ps[prow][ks * 32 + 8 * g],
                                 &Ps[prow][ks * 32 + 16 + 8 * g]);
            for (int j = 0; j < 2; j++) {
                int col = (nc * 2 + j) * 16 + cn;   // dh
                v16bf bv = load_frag(&Vt[col][ks * 32 + 8 * g],
                                     &Vt[col][ks * 32 + 16 + 8 * g]);
                oacc[j] = __builtin_amdgcn_wmma_f32_16x16x32_bf16(
                    false, pa, false, bv, (short)0, oacc[j], false, false);
            }
        }

        // wait for next chunk's async staging; also orders Vt/Ps/lrow reuse
        async_wait();
        __syncthreads();
    }

    // normalize and write O as [B,S,H*Dh] bf16 (GEMM-ready)
    for (int j = 0; j < 2; j++) {
        int dh = (nc * 2 + j) * 16 + cn;
        for (int r = 0; r < 8; r++) {
            int srow = qt * 64 + mi * 16 + r + 8 * g;
            float ov = oacc[j][r] / lrow[mi * 16 + r + 8 * g];
            O[(size_t)(b * NS + srow) * ND + h * NDH + dh] = (__bf16)ov;
        }
    }
}

// ---------------------------------------------------------------------------
// 4) Output projection: out[t, o] = sum_d O[t,d] * w_out[o,d] + b_out[o]  (f32)
// ---------------------------------------------------------------------------
__global__ __launch_bounds__(256) void out_gemm_kernel(
        const __bf16* __restrict__ X, const __bf16* __restrict__ W,
        const float* __restrict__ bias, float* __restrict__ out) {
    __shared__ __bf16 As[2][128][40];
    __shared__ __bf16 Bt[2][64][40];

    const int bm = blockIdx.x * 128, bn = blockIdx.y * 64;
    const int tid = threadIdx.x;
    const int w = tid >> 5, lane = tid & 31;
    const int g = lane >> 4, cn = lane & 15;
    const int wr = w >> 1, wc = w & 1;
    const int sr = tid >> 2, ss = (tid & 3) * 8;

    v8f acc[2][2] = {};

    cp16(&As[0][sr][ss],      &X[(size_t)(bm + sr) * ND + ss]);
    cp16(&As[0][sr + 64][ss], &X[(size_t)(bm + sr + 64) * ND + ss]);
    cp16(&Bt[0][sr][ss],      &W[(size_t)(bn + sr) * ND + ss]);
    async_wait();
    __syncthreads();

    for (int k0 = 0; k0 < ND; k0 += 32) {
        const int buf = (k0 >> 5) & 1;
        if (k0 + 32 < ND) {
            const int nk = k0 + 32, nb = buf ^ 1;
            cp16(&As[nb][sr][ss],      &X[(size_t)(bm + sr) * ND + nk + ss]);
            cp16(&As[nb][sr + 64][ss], &X[(size_t)(bm + sr + 64) * ND + nk + ss]);
            cp16(&Bt[nb][sr][ss],      &W[(size_t)(bn + sr) * ND + nk + ss]);
        }

        v16bf a[2], b[2];
        for (int i = 0; i < 2; i++) {
            int row = wr * 32 + i * 16 + cn;
            a[i] = load_frag(&As[buf][row][8 * g], &As[buf][row][16 + 8 * g]);
        }
        for (int j = 0; j < 2; j++) {
            int col = wc * 32 + j * 16 + cn;
            b[j] = load_frag(&Bt[buf][col][8 * g], &Bt[buf][col][16 + 8 * g]);
        }
        for (int i = 0; i < 2; i++)
            for (int j = 0; j < 2; j++)
                acc[i][j] = __builtin_amdgcn_wmma_f32_16x16x32_bf16(
                    false, a[i], false, b[j], (short)0, acc[i][j], false, false);

        async_wait();
        __syncthreads();
    }

    for (int i = 0; i < 2; i++)
        for (int j = 0; j < 2; j++) {
            int gcol = bn + wc * 32 + j * 16 + cn;
            float bb = bias[gcol];
            for (int r = 0; r < 8; r++) {
                int grow = bm + wr * 32 + i * 16 + r + 8 * g;
                out[(size_t)grow * ND + gcol] = acc[i][j][r] + bb;
            }
        }
}

// ---------------------------------------------------------------------------
// Launcher
// ---------------------------------------------------------------------------
extern "C" void kernel_launch(void* const* d_in, const int* in_sizes, int n_in,
                              void* d_out, int out_size, void* d_ws, size_t ws_size,
                              hipStream_t stream) {
    const float* x         = (const float*)d_in[0];
    const float* attn_bias = (const float*)d_in[1];
    const float* w_in      = (const float*)d_in[2];
    const float* b_in      = (const float*)d_in[3];
    const float* w_out     = (const float*)d_in[4];
    const float* b_out     = (const float*)d_in[5];
    float* out = (float*)d_out;

    char* ws = (char*)d_ws;
    const size_t SZ_X   = (size_t)NTOK * ND * 2;        // 8,388,608
    const size_t SZ_WIN = (size_t)N3D * ND * 2;         // 1,572,864
    const size_t SZ_WO  = (size_t)ND * ND * 2;          //   524,288
    const size_t SZ_QKV = (size_t)NB * NH * NS * NDH * 2;

    size_t off = 0;
    __bf16* xb    = (__bf16*)(ws + off); off += SZ_X;
    __bf16* w_inb = (__bf16*)(ws + off); off += SZ_WIN;
    __bf16* w_ob  = (__bf16*)(ws + off); off += SZ_WO;
    __bf16* Qb    = (__bf16*)(ws + off); off += SZ_QKV;
    __bf16* Kb    = (__bf16*)(ws + off); off += SZ_QKV;
    __bf16* Vb    = (__bf16*)(ws + off); off += SZ_QKV;
    __bf16* Ob    = (__bf16*)(ws + off); off += SZ_X;

    cvt_bf16_kernel<<<4096, 256, 0, stream>>>(x, xb, NTOK * ND);
    cvt_bf16_kernel<<<768, 256, 0, stream>>>(w_in, w_inb, N3D * ND);
    cvt_bf16_kernel<<<256, 256, 0, stream>>>(w_out, w_ob, ND * ND);

    qkv_gemm_kernel<<<dim3(NTOK / 128, N3D / 64), 256, 0, stream>>>(
        xb, w_inb, b_in, Qb, Kb, Vb);

    attn_kernel<<<dim3(NS / 64, NH, NB), 256, 0, stream>>>(
        Qb, Kb, Vb, attn_bias, Ob);

    out_gemm_kernel<<<dim3(NTOK / 128, ND / 64), 256, 0, stream>>>(
        Ob, w_ob, b_out, out);
}
